// Attention_65309272703072
// MI455X (gfx1250) — matrix operand order repaired and checked by reading, not compiled, not run on hardware
//
#include <hip/hip_runtime.h>
#include <hip/hip_bf16.h>
#include <math.h>

#define BB   2
#define DIM  256
#define NPTS 3072
#define HH   8
#define DH   32           // head dim == channels per group
#define BHN  (BB*HH)      // 16

// scale = 1/sqrt(32); work in log2 domain; prescale norms by CL2^2 so that
// p = exp2(-sqrt(CL2^2 * d2)) = exp(-sqrt(d2)*scale)
#define CL2   (0.17677669529663687f * 1.4426950408889634f)
#define CS2   (CL2 * CL2)
#define M2C   (-2.0f * CS2)
#define FLOORS (1e-12f * CS2)

typedef __attribute__((ext_vector_type(16))) _Float16 v16h;
typedef __attribute__((ext_vector_type(8)))  _Float16 v8h;
typedef __attribute__((ext_vector_type(8)))  float    v8f;

union FragH { v16h f; v8h h[2]; };
union PackH { _Float16 h[2]; unsigned u; };

// ---------------------------------------------------------------------------
// Kernel 1: grouped 1x1 conv -> Q,K,V (f16) + CL2^2*||q||^2, CL2^2*||k||^2
// grid (B*H, N/128), block 128 ; one thread per point n
// ---------------------------------------------------------------------------
__global__ __launch_bounds__(128) void qkv_kernel(
    const float* __restrict__ x, const float* __restrict__ wq,
    const float* __restrict__ wk, const float* __restrict__ wv,
    _Float16* __restrict__ Qg, _Float16* __restrict__ Kg, _Float16* __restrict__ Vg,
    float* __restrict__ q2g, float* __restrict__ k2g)
{
    __shared__ float wqs[DH][DH], wks[DH][DH], wvs[DH][DH];
    const int bh = blockIdx.x;
    const int b  = bh >> 3, h = bh & 7;
    const int n  = blockIdx.y * 128 + threadIdx.x;

    for (int i = threadIdx.x; i < DH * DH; i += 128) {
        wqs[i >> 5][i & 31] = wq[h * DH * DH + i];
        wks[i >> 5][i & 31] = wk[h * DH * DH + i];
        wvs[i >> 5][i & 31] = wv[h * DH * DH + i];
    }
    __syncthreads();

    float xv[DH];
#pragma unroll
    for (int c = 0; c < DH; ++c)
        xv[c] = x[((size_t)b * DIM + h * DH + c) * NPTS + n];

    float q2 = 0.f, k2 = 0.f;
    const size_t obase = ((size_t)bh * NPTS + n) * DH;
#pragma unroll 4
    for (int d = 0; d < DH; ++d) {
        float aq = 0.f, ak = 0.f, av = 0.f;
#pragma unroll
        for (int c = 0; c < DH; ++c) {
            aq = fmaf(xv[c], wqs[d][c], aq);
            ak = fmaf(xv[c], wks[d][c], ak);
            av = fmaf(xv[c], wvs[d][c], av);
        }
        q2 += aq * aq;
        k2 += ak * ak;
        Qg[obase + d] = (_Float16)aq;
        Kg[obase + d] = (_Float16)ak;
        Vg[obase + d] = (_Float16)av;
    }
    q2g[(size_t)bh * NPTS + n] = q2 * CS2;   // prescaled into log2 domain
    k2g[(size_t)bh * NPTS + n] = k2 * CS2;
}

// ---------------------------------------------------------------------------
// Kernel 2: flash attention with L2 similarity, f16 WMMA + f32 accumulation.
// Scores are always <= 0 -> exp <= 1, row sum <= N: no online max needed.
// Chunk keys are interleaved: S-fragment 0 = even keys, fragment 1 = odd keys,
// so each lane's two p values are adjacent in Ps -> one packed b32 store.
// Softmax denominators are accumulated by the matrix unit: lacc += P x ones.
// grid (B*H, N/128), block 256 (8 waves x 16 query rows)
// ---------------------------------------------------------------------------
__global__ __launch_bounds__(256) void attn_kernel(
    const _Float16* __restrict__ Qg, const _Float16* __restrict__ Kg,
    const _Float16* __restrict__ Vg, const float* __restrict__ q2g,
    const float* __restrict__ k2g, float* __restrict__ Og)
{
    // LDS strides of 40 halves = 80 B (multiple of 16 B -> b128-aligned rows)
    __shared__ _Float16 Ks[32][40];        // Ks[key][d]
    __shared__ _Float16 Vs[32][40];        // Vs[d][key]   (transposed)
    __shared__ float    k2s[32];
    __shared__ _Float16 Ps[8][16][40];     // per-wave softmax tile [row][key]

    const int bh   = blockIdx.x;
    const int m0   = blockIdx.y * 128;
    const int tid  = threadIdx.x;
    const int lane = tid & 31;
    const int wave = tid >> 5;
    const int mlo  = lane & 15;
    const int hi   = lane >> 4;

    const _Float16* Qb = Qg + (size_t)bh * NPTS * DH;
    const _Float16* Kb = Kg + (size_t)bh * NPTS * DH;
    const _Float16* Vb = Vg + (size_t)bh * NPTS * DH;

    // ---- Q A-fragment (16x32): lane = row, K = [16hi,16hi+16) contiguous
    FragH qa;
    {
        const _Float16* qrow = Qb + (size_t)(m0 + wave * 16 + mlo) * DH;
        qa.h[0] = *(const v8h*)(qrow + 16 * hi);
        qa.h[1] = *(const v8h*)(qrow + 16 * hi + 8);
    }
    float q2v[8];
#pragma unroll
    for (int r = 0; r < 8; ++r)
        q2v[r] = q2g[(size_t)bh * NPTS + m0 + wave * 16 + r + 8 * hi];

    // all-ones B fragment: row_sum(P) = P x ones (layout-independent)
    v16h ones;
#pragma unroll
    for (int i = 0; i < 16; ++i) ones[i] = (_Float16)1.0f;

    v8f o0   = {0.f, 0.f, 0.f, 0.f, 0.f, 0.f, 0.f, 0.f};
    v8f o1   = o0;
    v8f lacc = o0;                          // softmax denominators (all cols equal)

    for (int kc = 0; kc < NPTS / 32; ++kc) {
        __syncthreads();
        // stage K chunk: 32 keys x 32 d, 8B per thread
        const int key = tid >> 3, dg = (tid & 7) * 4;
        *(uint2*)&Ks[key][dg] =
            *(const uint2*)(Kb + ((size_t)(kc * 32 + key)) * DH + dg);
        // stage V chunk transposed: one 8B load, four half stores
        {
            PackH vv[2];
            *(uint2*)vv = *(const uint2*)(Vb + ((size_t)(kc * 32 + key)) * DH + dg);
            Vs[dg + 0][key] = vv[0].h[0];
            Vs[dg + 1][key] = vv[0].h[1];
            Vs[dg + 2][key] = vv[1].h[0];
            Vs[dg + 3][key] = vv[1].h[1];
        }
        if (tid < 32) k2s[tid] = k2g[(size_t)bh * NPTS + kc * 32 + tid];
        // prefetch next chunk (global_prefetch_b8)
        if (kc + 1 < NPTS / 32 && tid < 32) {
            __builtin_prefetch(Kb + ((size_t)((kc + 1) * 32 + tid)) * DH, 0, 1);
            __builtin_prefetch(Vb + ((size_t)((kc + 1) * 32 + tid)) * DH, 0, 1);
        }
        __syncthreads();

        // ---- S = Q * K^T ; fragment 0 covers even chunk keys, fragment 1 odd
        FragH kb0, kb1;
        kb0.h[0] = *(const v8h*)&Ks[2 * mlo][16 * hi];
        kb0.h[1] = *(const v8h*)&Ks[2 * mlo][16 * hi + 8];
        kb1.h[0] = *(const v8h*)&Ks[2 * mlo + 1][16 * hi];
        kb1.h[1] = *(const v8h*)&Ks[2 * mlo + 1][16 * hi + 8];

        v8f s0 = {0.f,0.f,0.f,0.f,0.f,0.f,0.f,0.f};
        v8f s1 = s0;
        s0 = __builtin_amdgcn_wmma_f32_16x16x32_f16(false, qa.f, false, kb0.f,
                                                    (short)0, s0, false, false);
        s1 = __builtin_amdgcn_wmma_f32_16x16x32_f16(false, qa.f, false, kb1.f,
                                                    (short)0, s1, false, false);

        const float2 k2p = *(const float2*)&k2s[2 * mlo];   // even/odd key norms

        // ---- p = exp2(-sqrt(d2')), everything prescaled by CL2^2
#pragma unroll
        for (int r = 0; r < 8; ++r) {
            float d0 = fmaxf(fmaf(s0[r], M2C, q2v[r] + k2p.x), FLOORS);
            float d1 = fmaxf(fmaf(s1[r], M2C, q2v[r] + k2p.y), FLOORS);
            float p0 = __builtin_amdgcn_exp2f(-__builtin_amdgcn_sqrtf(d0));
            float p1 = __builtin_amdgcn_exp2f(-__builtin_amdgcn_sqrtf(d1));
            PackH pk;
            pk.h[0] = (_Float16)p0;
            pk.h[1] = (_Float16)p1;
            *(unsigned*)&Ps[wave][r + 8 * hi][2 * mlo] = pk.u;
        }
        // same-wave LDS ops are in-order; just stop the compiler reordering
        asm volatile("" ::: "memory");

        // ---- P A-fragment + V^T B-fragments, accumulate O and denominators
        FragH pa, vb0, vb1;
        pa.h[0]  = *(const v8h*)&Ps[wave][mlo][16 * hi];
        pa.h[1]  = *(const v8h*)&Ps[wave][mlo][16 * hi + 8];
        vb0.h[0] = *(const v8h*)&Vs[mlo][16 * hi];        // d = mlo,    j = e+16hi
        vb0.h[1] = *(const v8h*)&Vs[mlo][16 * hi + 8];
        vb1.h[0] = *(const v8h*)&Vs[16 + mlo][16 * hi];   // d = 16+mlo
        vb1.h[1] = *(const v8h*)&Vs[16 + mlo][16 * hi + 8];

        o0   = __builtin_amdgcn_wmma_f32_16x16x32_f16(false, pa.f, false, vb0.f,
                                                      (short)0, o0, false, false);
        o1   = __builtin_amdgcn_wmma_f32_16x16x32_f16(false, pa.f, false, vb1.f,
                                                      (short)0, o1, false, false);
        lacc = __builtin_amdgcn_wmma_f32_16x16x32_f16(false, pa.f, false, ones,
                                                      (short)0, lacc, false, false);
    }

    // ---- every lane already holds its rows' full denominators in lacc
#pragma unroll
    for (int r = 0; r < 8; ++r) {
        const float inv = __builtin_amdgcn_rcpf(lacc[r]);
        const int row = m0 + wave * 16 + r + 8 * hi;
        Og[((size_t)bh * NPTS + row) * DH + mlo]      = o0[r] * inv;
        Og[((size_t)bh * NPTS + row) * DH + 16 + mlo] = o1[r] * inv;
    }
}

// ---------------------------------------------------------------------------
// Kernel 3: grouped output projection + bias -> y (B, DIM, N)
// ---------------------------------------------------------------------------
__global__ __launch_bounds__(128) void proj_kernel(
    const float* __restrict__ Og, const float* __restrict__ wo,
    const float* __restrict__ bo, float* __restrict__ y)
{
    __shared__ float wos[DH][DH];   // wos[c][d]
    const int bh = blockIdx.x;
    const int b  = bh >> 3, h = bh & 7;
    const int n  = blockIdx.y * 128 + threadIdx.x;

    for (int i = threadIdx.x; i < DH * DH; i += 128)
        wos[i >> 5][i & 31] = wo[h * DH * DH + i];
    __syncthreads();

    float ov[DH];
#pragma unroll
    for (int d = 0; d < DH; ++d)
        ov[d] = Og[((size_t)bh * NPTS + n) * DH + d];

#pragma unroll 4
    for (int c = 0; c < DH; ++c) {
        float acc = bo[h * DH + c];
#pragma unroll
        for (int d = 0; d < DH; ++d)
            acc = fmaf(ov[d], wos[c][d], acc);
        y[((size_t)b * DIM + h * DH + c) * NPTS + n] = acc;
    }
}

// ---------------------------------------------------------------------------
// Kernel 4: per-channel batch stats (mean, rsqrt(var+eps)) over (B, N)
// ---------------------------------------------------------------------------
__global__ __launch_bounds__(256) void stats_kernel(
    const float* __restrict__ y, float* __restrict__ meanp, float* __restrict__ rstdp)
{
    const int c = blockIdx.x;
    float s = 0.f, ss = 0.f;
    for (int i = threadIdx.x; i < BB * NPTS; i += 256) {
        const int b = i / NPTS, n = i - b * NPTS;
        const float v = y[((size_t)b * DIM + c) * NPTS + n];
        s += v; ss += v * v;
    }
#pragma unroll
    for (int off = 16; off >= 1; off >>= 1) {
        s  += __shfl_xor(s,  off, 32);
        ss += __shfl_xor(ss, off, 32);
    }
    __shared__ float sb[8], ssb[8];
    if ((threadIdx.x & 31) == 0) { sb[threadIdx.x >> 5] = s; ssb[threadIdx.x >> 5] = ss; }
    __syncthreads();
    if (threadIdx.x == 0) {
        float S = 0.f, SS = 0.f;
#pragma unroll
        for (int w = 0; w < 8; ++w) { S += sb[w]; SS += ssb[w]; }
        const float inv = 1.f / (float)(BB * NPTS);
        const float mu  = S * inv;
        const float var = fmaxf(SS * inv - mu * mu, 0.f);
        meanp[c] = mu;
        rstdp[c] = rsqrtf(var + 1e-5f);
    }
}

// ---------------------------------------------------------------------------
// Kernel 5: apply batchnorm affine -> out
// ---------------------------------------------------------------------------
__global__ __launch_bounds__(256) void bn_kernel(
    const float* __restrict__ y, const float* __restrict__ meanp,
    const float* __restrict__ rstdp, const float* __restrict__ gamma,
    const float* __restrict__ beta, float* __restrict__ out)
{
    const size_t i = (size_t)blockIdx.x * 256 + threadIdx.x;
    if (i >= (size_t)BB * DIM * NPTS) return;
    const int c = (int)((i / NPTS) % DIM);
    out[i] = (y[i] - meanp[c]) * rstdp[c] * gamma[c] + beta[c];
}

// ---------------------------------------------------------------------------
extern "C" void kernel_launch(void* const* d_in, const int* in_sizes, int n_in,
                              void* d_out, int out_size, void* d_ws, size_t ws_size,
                              hipStream_t stream)
{
    const float* x     = (const float*)d_in[0];
    const float* wq    = (const float*)d_in[1];
    const float* wk    = (const float*)d_in[2];
    const float* wv    = (const float*)d_in[3];
    const float* wo    = (const float*)d_in[4];
    const float* bo    = (const float*)d_in[5];
    const float* gamma = (const float*)d_in[6];
    const float* beta  = (const float*)d_in[7];
    float* out = (float*)d_out;

    char* ws = (char*)d_ws;
    const size_t nqkv = (size_t)BHN * NPTS * DH;            // 1,572,864 elems
    _Float16* Qg  = (_Float16*)(ws);                        // 2B each
    _Float16* Kg  = (_Float16*)(ws + nqkv * 2);
    _Float16* Vg  = (_Float16*)(ws + nqkv * 4);
    float*    q2g = (float*)(ws + nqkv * 6);
    float*    k2g = (float*)(ws + nqkv * 6 + (size_t)BHN * NPTS * 4);
    float*    Og  = (float*)(ws + nqkv * 6 + (size_t)BHN * NPTS * 8);
    float*    yb  = (float*)((char*)Og + nqkv * 4);
    float*    mp  = (float*)((char*)yb + (size_t)BB * DIM * NPTS * 4);
    float*    rp  = mp + DIM;

    qkv_kernel  <<<dim3(BHN, NPTS / 128), 128, 0, stream>>>(x, wq, wk, wv,
                                                            Qg, Kg, Vg, q2g, k2g);
    attn_kernel <<<dim3(BHN, NPTS / 128), 256, 0, stream>>>(Qg, Kg, Vg, q2g, k2g, Og);
    proj_kernel <<<dim3(BHN, NPTS / 128), 128, 0, stream>>>(Og, wo, bo, yb);
    stats_kernel<<<DIM, 256, 0, stream>>>(yb, mp, rp);
    bn_kernel   <<<(BB * DIM * NPTS + 255) / 256, 256, 0, stream>>>(yb, mp, rp,
                                                                    gamma, beta, out);
}
